// SimpleMoEModel_44092134260789
// MI455X (gfx1250) — matrix operand order repaired and checked by reading, not compiled, not run on hardware
//
#include <hip/hip_runtime.h>
#include <math.h>

// ---- problem constants (match reference) ----
#define B_    4
#define S_    4096
#define H_    512
#define E_    8
#define FF_   1024
#define NCLS_ 10
#define T_    (B_ * S_)     // 16384 tokens
#define MTILE 64            // tokens per block

typedef __attribute__((ext_vector_type(16))) _Float16 v16h;
typedef __attribute__((ext_vector_type(8)))  _Float16 v8h;
typedef __attribute__((ext_vector_type(8)))  float    v8f;

// ---------------------------------------------------------------------------
// WMMA fragment loaders (contiguous k order used by this code for A and B, wave32)
// A (16x32, MxK): lanes 0-15 row M=lane, K=0..15 contiguous;
//                 lanes 16-31 row M=lane-16, K=16..31.
// B (32x16, KxN) from N-major (transposed) memory: lanes 0-15 col N=lane,
//                 K=0..15 contiguous; lanes 16-31 col N=lane-16, K=16..31.
// ---------------------------------------------------------------------------
__device__ __forceinline__ v16h load_a16(const _Float16* base, int ld, int lane) {
  const _Float16* p = base + (size_t)(lane & 15) * ld + ((lane >> 4) << 4);
  v16h a;
  ((v8h*)&a)[0] = *(const v8h*)(p);        // K +0..7  (or +16..23)
  ((v8h*)&a)[1] = *(const v8h*)(p + 8);    // K +8..15 (or +24..31)
  return a;
}

__device__ __forceinline__ v16h load_b16(const _Float16* base, int ld, int lane) {
  // base points at [N-major] tile origin; row stride ld = K extent of source
  const _Float16* p = base + (size_t)(lane & 15) * ld + ((lane >> 4) << 4);
  return *(const v16h*)(p);                // 16 contiguous K values
}

__device__ __forceinline__ v8f wmma16(v16h a, v16h b, v8f c) {
  return __builtin_amdgcn_wmma_f32_16x16x32_f16(false, a, false, b, (short)0, c,
                                                false, false);
}

// ---------------------------------------------------------------------------
// fp32 -> fp16 conversion kernels (one-time per launch; tiny vs GEMM cost)
// ---------------------------------------------------------------------------
__global__ void cvt_x_kernel(const float* __restrict__ x, _Float16* __restrict__ xh, int n) {
  for (int i = blockIdx.x * blockDim.x + threadIdx.x; i < n; i += gridDim.x * blockDim.x)
    xh[i] = (_Float16)x[i];
}

// w1[E,H,FF] -> w1t[E,FF,H]  (N-major for GEMM1 B-fragments)
__global__ void cvt_w1t_kernel(const float* __restrict__ w1, _Float16* __restrict__ w1t) {
  const int n = E_ * FF_ * H_;
  for (int i = blockIdx.x * blockDim.x + threadIdx.x; i < n; i += gridDim.x * blockDim.x) {
    int e = i / (FF_ * H_);
    int r = i - e * (FF_ * H_);
    int f = r / H_;
    int h = r - f * H_;
    w1t[i] = (_Float16)w1[(size_t)e * H_ * FF_ + (size_t)h * FF_ + f];
  }
}

// w2[E,FF,H] -> w2t[E,H,FF]  (N-major for GEMM2 B-fragments)
__global__ void cvt_w2t_kernel(const float* __restrict__ w2, _Float16* __restrict__ w2t) {
  const int n = E_ * H_ * FF_;
  for (int i = blockIdx.x * blockDim.x + threadIdx.x; i < n; i += gridDim.x * blockDim.x) {
    int e = i / (H_ * FF_);
    int r = i - e * (H_ * FF_);
    int h = r / FF_;
    int f = r - h * FF_;
    w2t[i] = (_Float16)w2[(size_t)e * FF_ * H_ + (size_t)f * H_ + h];
  }
}

// ---------------------------------------------------------------------------
// Router: softmax(x @ router_w + router_b) in fp32 (tiny: 0.13 GFLOP)
// block = 256 threads = 32 tokens x 8 experts
// ---------------------------------------------------------------------------
__global__ void router_kernel(const float* __restrict__ x,
                              const float* __restrict__ rwgt,
                              const float* __restrict__ rbias,
                              float* __restrict__ rw) {
  __shared__ float lg[32][E_];
  const int tl = threadIdx.x >> 3;
  const int e  = threadIdx.x & 7;
  const int t  = blockIdx.x * 32 + tl;
  const float* xr = x + (size_t)t * H_;
  float acc = rbias[e];
  for (int h = 0; h < H_; ++h) acc += xr[h] * rwgt[h * E_ + e];
  lg[tl][e] = acc;
  __syncthreads();
  if (e == 0) {
    float mx = lg[tl][0];
    #pragma unroll
    for (int i = 1; i < E_; ++i) mx = fmaxf(mx, lg[tl][i]);
    float ex[E_]; float sum = 0.0f;
    #pragma unroll
    for (int i = 0; i < E_; ++i) { ex[i] = expf(lg[tl][i] - mx); sum += ex[i]; }
    const float inv = 1.0f / sum;
    #pragma unroll
    for (int i = 0; i < E_; ++i) rw[(size_t)t * E_ + i] = ex[i] * inv;
  }
}

// ---------------------------------------------------------------------------
// Fused MoE: per 64-token tile, loop experts:
//   h1 = gelu(x @ w1 + b1)           (WMMA, f32 acc, h1 -> LDS as f16)
//   out += rw_e * (h1 @ w2 + b2)     (WMMA, f32 acc, combine in LDS f32)
// then mean-pool rows into pooled[B,H] via atomicAdd.
// LDS: out_lds 64*512*4 = 128KB + h1_lds 64*1024*2 = 128KB -> 256KB dynamic.
// 8 waves: wave -> (row-tile rt = wv>>1, column half chv = wv&1).
// ---------------------------------------------------------------------------
__global__ __launch_bounds__(256, 1)
void moe_expert_kernel(const _Float16* __restrict__ xh,
                       const _Float16* __restrict__ w1t,
                       const _Float16* __restrict__ w2t,
                       const float* __restrict__ b1,
                       const float* __restrict__ b2,
                       const float* __restrict__ rw,
                       float* __restrict__ pooled) {
  extern __shared__ char smem[];
  float*    out_lds = (float*)smem;                                  // [64][512]
  _Float16* h1_lds  = (_Float16*)(smem + MTILE * H_ * sizeof(float)); // [64][1024]

  const int tid   = threadIdx.x;
  const int lane  = tid & 31;
  const int wv    = tid >> 5;
  const int rt    = wv >> 1;   // 0..3  (row tile of 16 tokens)
  const int chv   = wv & 1;    // 0..1  (column half)
  const int m0    = blockIdx.x * MTILE;
  const int batch = m0 / S_;
  const int hi8   = (lane >> 4) << 3;  // accumulator row offset (0 or 8)

  for (int i = tid; i < MTILE * H_; i += 256) out_lds[i] = 0.0f;

  for (int e = 0; e < E_; ++e) {
    __syncthreads();  // prev GEMM2 done reading h1 / zero-init visible
    const _Float16* w1e = w1t + (size_t)e * FF_ * H_;
    const _Float16* w2e = w2t + (size_t)e * H_ * FF_;
    const float*    b1e = b1 + e * FF_;
    const float*    b2e = b2 + e * H_;
    const _Float16* xa  = xh + (size_t)(m0 + rt * 16) * H_;

    // ---- GEMM1: rows rt*16..+15, FF cols [chv*512, chv*512+512) ----
    for (int g = 0; g < 8; ++g) {
      const int ct0 = chv * 32 + g * 4;     // first of 4 FF col-tiles
      v8f acc[4] = {};
      for (int k = 0; k < H_; k += 32) {
        v16h a = load_a16(xa + k, H_, lane);
        #pragma unroll
        for (int u = 0; u < 4; ++u) {
          v16h bf = load_b16(w1e + (size_t)(ct0 + u) * 16 * H_ + k, H_, lane);
          acc[u] = wmma16(a, bf, acc[u]);
        }
      }
      #pragma unroll
      for (int u = 0; u < 4; ++u) {
        const int n = (ct0 + u) * 16 + (lane & 15);
        const float b1v = b1e[n];
        #pragma unroll
        for (int j = 0; j < 8; ++j) {
          const int ml = rt * 16 + j + hi8;
          float v = acc[u][j] + b1v;
          v = 0.5f * v * (1.0f + erff(v * 0.70710678118654752f));  // exact gelu
          h1_lds[ml * FF_ + n] = (_Float16)v;
        }
      }
    }
    __syncthreads();  // h1 complete

    // router weights for this wave's 8 accumulator rows
    float rwl[8];
    #pragma unroll
    for (int j = 0; j < 8; ++j)
      rwl[j] = rw[(size_t)(m0 + rt * 16 + j + hi8) * E_ + e];

    // ---- GEMM2: rows rt*16..+15, H cols [chv*256, chv*256+256) ----
    for (int g = 0; g < 4; ++g) {
      const int ct0 = chv * 16 + g * 4;     // first of 4 H col-tiles
      v8f acc[4] = {};
      for (int k = 0; k < FF_; k += 32) {
        v16h a = load_a16(&h1_lds[(rt * 16) * FF_ + k], FF_, lane);
        #pragma unroll
        for (int u = 0; u < 4; ++u) {
          v16h bf = load_b16(w2e + (size_t)(ct0 + u) * 16 * FF_ + k, FF_, lane);
          acc[u] = wmma16(a, bf, acc[u]);
        }
      }
      #pragma unroll
      for (int u = 0; u < 4; ++u) {
        const int n = (ct0 + u) * 16 + (lane & 15);
        const float b2v = b2e[n];
        #pragma unroll
        for (int j = 0; j < 8; ++j) {
          const int ml = rt * 16 + j + hi8;
          out_lds[ml * H_ + n] += rwl[j] * (acc[u][j] + b2v);  // owner-wave only
        }
      }
    }
  }
  __syncthreads();

  // mean-pool over the 64 rows (all same batch since S % MTILE == 0)
  for (int c = tid; c < H_; c += 256) {
    float s = 0.0f;
    for (int m = 0; m < MTILE; ++m) s += out_lds[m * H_ + c];
    atomicAdd(&pooled[batch * H_ + c], s * (1.0f / (float)S_));
  }
}

// ---------------------------------------------------------------------------
// Classifier: logits[b,c] = pooled[b,:] @ clf_w[:,c] + clf_b[c]   (40 outputs)
// ---------------------------------------------------------------------------
__global__ void clf_kernel(const float* __restrict__ pooled,
                           const float* __restrict__ cw,
                           const float* __restrict__ cb,
                           float* __restrict__ out) {
  const int t = threadIdx.x;
  if (t < B_ * NCLS_) {
    const int b = t / NCLS_, c = t - b * NCLS_;
    float s = cb[c];
    for (int h = 0; h < H_; ++h) s += pooled[b * H_ + h] * cw[h * NCLS_ + c];
    out[t] = s;
  }
}

// ---------------------------------------------------------------------------
// Workspace layout (bytes):
//   xh  : f16 [T,H]        @ 0          (16,777,216)
//   w1t : f16 [E,FF,H]     @ 16777216   ( 8,388,608)
//   w2t : f16 [E,H,FF]     @ 25165824   ( 8,388,608)
//   rw  : f32 [T,E]        @ 33554432   (   524,288)
//   pooled: f32 [B,H]      @ 34078720   (     8,192)
// ---------------------------------------------------------------------------
extern "C" void kernel_launch(void* const* d_in, const int* in_sizes, int n_in,
                              void* d_out, int out_size, void* d_ws, size_t ws_size,
                              hipStream_t stream) {
  const float* x        = (const float*)d_in[0];
  const float* router_w = (const float*)d_in[1];
  const float* router_b = (const float*)d_in[2];
  const float* w1       = (const float*)d_in[3];
  const float* b1       = (const float*)d_in[4];
  const float* w2       = (const float*)d_in[5];
  const float* b2       = (const float*)d_in[6];
  const float* clf_w    = (const float*)d_in[7];
  const float* clf_b    = (const float*)d_in[8];
  float* out = (float*)d_out;

  char* ws = (char*)d_ws;
  _Float16* xh     = (_Float16*)(ws + 0);
  _Float16* w1t    = (_Float16*)(ws + 16777216);
  _Float16* w2t    = (_Float16*)(ws + 25165824);
  float*    rw     = (float*)   (ws + 33554432);
  float*    pooled = (float*)   (ws + 34078720);

  hipMemsetAsync(pooled, 0, B_ * H_ * sizeof(float), stream);

  cvt_x_kernel  <<<2048, 256, 0, stream>>>(x, xh, T_ * H_);
  cvt_w1t_kernel<<<2048, 256, 0, stream>>>(w1, w1t);
  cvt_w2t_kernel<<<2048, 256, 0, stream>>>(w2, w2t);
  router_kernel <<<T_ / 32, 256, 0, stream>>>(x, router_w, router_b, rw);

  const int smem_bytes = MTILE * H_ * (int)sizeof(float)      // out_lds 128KB
                       + MTILE * FF_ * (int)sizeof(_Float16); // h1_lds 128KB
  hipFuncSetAttribute(reinterpret_cast<const void*>(moe_expert_kernel),
                      hipFuncAttributeMaxDynamicSharedMemorySize, smem_bytes);
  moe_expert_kernel<<<T_ / MTILE, 256, smem_bytes, stream>>>(xh, w1t, w2t, b1, b2,
                                                             rw, pooled);

  clf_kernel<<<1, 64, 0, stream>>>(pooled, clf_w, clf_b, out);
}
